// MoeExperts_22986664968196
// MI455X (gfx1250) — compile-verified, matrix operand order checked by reading, not run on hardware
//
#include <hip/hip_runtime.h>
#include <hip/hip_bf16.h>
#include <math.h>

#define E_EXPERTS 32
#define HDIM      1024
#define IDIM      768
#define TTOK      4096
#define TOPK      4
#define CAPQ      1024
#define NASSIGN   (TTOK * TOPK)

typedef __attribute__((ext_vector_type(16))) __bf16 v16bf;
typedef __attribute__((ext_vector_type(8)))  float  v8f;

union Frag {
    v16bf v;
    uint4 q[2];
};

__device__ __forceinline__ unsigned short f32_to_bf16_rne(float f) {
    unsigned int u = __float_as_uint(f);
    return (unsigned short)((u + 0x7FFFu + ((u >> 16) & 1u)) >> 16);
}

// pack two fp32 -> two bf16 in one dword (lo in [15:0], hi in [31:16]), RNE
__device__ __forceinline__ unsigned int bf16_pack2(float lo, float hi) {
    unsigned int ul = __float_as_uint(lo);
    unsigned int uh = __float_as_uint(hi);
    ul = (ul + 0x7FFFu + ((ul >> 16) & 1u)) >> 16;
    uh = (uh + 0x7FFFu + ((uh >> 16) & 1u)) & 0xFFFF0000u;
    return ul | uh;
}

// ---------------------------------------------------------------- utilities
__global__ void zero_f32_kernel(float* __restrict__ p, int n) {
    for (int i = blockIdx.x * blockDim.x + threadIdx.x; i < n;
         i += gridDim.x * blockDim.x)
        p[i] = 0.0f;
}

__global__ void zero_i32_kernel(int* __restrict__ p, int n) {
    int i = blockIdx.x * blockDim.x + threadIdx.x;
    if (i < n) p[i] = 0;
}

// one-shot streaming fp32 -> bf16 conversion (float4 in, uint2 out)
__global__ void cvt_f32_bf16_kernel(const float* __restrict__ src,
                                    unsigned int* __restrict__ dst, int n4) {
    for (int i = blockIdx.x * blockDim.x + threadIdx.x; i < n4;
         i += gridDim.x * blockDim.x) {
        float4 x = ((const float4*)src)[i];
        uint2 o;
        o.x = bf16_pack2(x.x, x.y);
        o.y = bf16_pack2(x.z, x.w);
        ((uint2*)dst)[i] = o;
    }
}

// ------------------------------------------------------------- routing scatter
__global__ void route_scatter_kernel(const int* __restrict__ topk_idx,
                                     int* __restrict__ counts,
                                     int* __restrict__ slots) {
    int a = blockIdx.x * blockDim.x + threadIdx.x;
    if (a >= NASSIGN) return;
    int e = topk_idx[a];
    int pos = atomicAdd(&counts[e], 1);
    if (pos < CAPQ) slots[e * CAPQ + pos] = a;
}

// ---------------------------------------------------- GEMM A: gate_up + SwiGLU
// Block: 256 thr (8 waves, 2Mx4N). Tile: 32 tokens x 64 I-cols. K-step 64.
// Double-buffered LDS, one barrier per K-chunk.
__global__ __launch_bounds__(256) void moe_gateup_kernel(
    const unsigned short* __restrict__ hsb,   // [T, H] bf16
    const unsigned short* __restrict__ gupb,  // [E, 2I, H] bf16
    const int*   __restrict__ counts,
    const int*   __restrict__ slots,
    unsigned short* __restrict__ hbuf)        // [NASSIGN, I] bf16
{
    const int e  = blockIdx.z;
    const int m0 = blockIdx.y * 32;
    const int j0 = blockIdx.x * 64;
    const int cnt = counts[e];
    if (m0 >= cnt) return;

    __shared__ unsigned short sA[2][32][72];   // 32 tok x 64 k (+8 pad)
    __shared__ unsigned short sB[2][128][72];  // 64 gate + 64 up rows
    __shared__ int sSlot[32];

    const int tid  = threadIdx.x;
    const int lane = tid & 31;
    const int wave = tid >> 5;
    const int wm   = wave >> 2;
    const int wn   = wave & 3;

    if (tid < 32) {
        int r = m0 + tid;
        sSlot[tid] = (r < cnt) ? slots[e * CAPQ + r] : -1;
    }
    __syncthreads();

    // hoisted, branch-free staging sources (invalid rows read token 0;
    // their outputs land only in masked-off rows)
    const int rA = tid >> 3;
    const int cA = (tid & 7) * 8;
    const int slotA = sSlot[rA];
    const unsigned short* aSrc =
        hsb + (size_t)((slotA >= 0) ? (slotA >> 2) : 0) * HDIM + cA;

    const unsigned short* wg = gupb + (size_t)e * (2 * IDIM) * HDIM;
    const unsigned short* bSrc[4];
    int bR[4], bC[4];
#pragma unroll
    for (int i = 0; i < 4; ++i) {
        int f  = tid + 256 * i;           // 0..1023 (8-half chunks)
        bR[i]  = f >> 3;                  // 0..127
        bC[i]  = (f & 7) * 8;
        int wrow = (bR[i] < 64) ? (j0 + bR[i]) : (IDIM + j0 + (bR[i] - 64));
        bSrc[i] = wg + (size_t)wrow * HDIM + bC[i];
    }

    // prologue: stage chunk 0 into buffer 0
    {
        *(uint4*)&sA[0][rA][cA] = *(const uint4*)aSrc;
#pragma unroll
        for (int i = 0; i < 4; ++i)
            *(uint4*)&sB[0][bR[i]][bC[i]] = *(const uint4*)bSrc[i];
    }
    __syncthreads();

    v8f cg = {};
    v8f cu = {};
    const int kb   = (lane >> 4) * 8;
    const int mrow = wm * 16 + (lane & 15);
    const int nrow = wn * 16 + (lane & 15);

    int cur = 0;
    for (int k0 = 0; k0 < HDIM; k0 += 64) {
        const int kn = k0 + 64;
        const bool more = kn < HDIM;
        uint4 ra, rb0, rb1, rb2, rb3;
        if (more) {                       // prefetch next chunk into regs
            ra  = *(const uint4*)(aSrc + kn);
            rb0 = *(const uint4*)(bSrc[0] + kn);
            rb1 = *(const uint4*)(bSrc[1] + kn);
            rb2 = *(const uint4*)(bSrc[2] + kn);
            rb3 = *(const uint4*)(bSrc[3] + kn);
        }
        // compute on current buffer
#pragma unroll
        for (int kk = 0; kk < 64; kk += 32) {
            Frag fa, fg, fu;
            fa.q[0] = *(const uint4*)&sA[cur][mrow][kk + kb];
            fa.q[1] = *(const uint4*)&sA[cur][mrow][kk + kb + 16];
            fg.q[0] = *(const uint4*)&sB[cur][nrow][kk + kb];
            fg.q[1] = *(const uint4*)&sB[cur][nrow][kk + kb + 16];
            fu.q[0] = *(const uint4*)&sB[cur][64 + nrow][kk + kb];
            fu.q[1] = *(const uint4*)&sB[cur][64 + nrow][kk + kb + 16];
            cg = __builtin_amdgcn_wmma_f32_16x16x32_bf16(
                false, fa.v, false, fg.v, (short)0, cg, false, false);
            cu = __builtin_amdgcn_wmma_f32_16x16x32_bf16(
                false, fa.v, false, fu.v, (short)0, cu, false, false);
        }
        if (more) {                       // store prefetch into other buffer
            int nxt = cur ^ 1;
            *(uint4*)&sA[nxt][rA][cA] = ra;
            *(uint4*)&sB[nxt][bR[0]][bC[0]] = rb0;
            *(uint4*)&sB[nxt][bR[1]][bC[1]] = rb1;
            *(uint4*)&sB[nxt][bR[2]][bC[2]] = rb2;
            *(uint4*)&sB[nxt][bR[3]][bC[3]] = rb3;
        }
        __syncthreads();
        cur ^= 1;
    }

    // epilogue: SwiGLU via fast rcp; store h row = assignment id
    const int ncol = j0 + wn * 16 + (lane & 15);
    const int mhi  = (lane >> 4) * 8;
#pragma unroll
    for (int r = 0; r < 8; ++r) {
        int mi = wm * 16 + mhi + r;
        if (m0 + mi < cnt) {
            int a = sSlot[mi];
            float g = cg[r];
            float u = cu[r];
            float sig = __builtin_amdgcn_rcpf(1.0f + __expf(-g));
            hbuf[(size_t)a * IDIM + ncol] = f32_to_bf16_rne(g * sig * u);
        }
    }
}

// ------------------------------------------------ GEMM B: down proj + scatter
// Block: 256 thr (8 waves, 2Mx4N). Tile: 32 rows x 64 H-cols. K-step 64.
__global__ __launch_bounds__(256) void moe_down_kernel(
    const unsigned short* __restrict__ hbuf,  // [NASSIGN, I] bf16
    const unsigned short* __restrict__ dwnb,  // [E, H, I] bf16
    const float* __restrict__ topk_w,         // [T*K] fp32
    const int*   __restrict__ counts,
    const int*   __restrict__ slots,
    float* __restrict__ y)                    // [T, H] fp32
{
    const int e  = blockIdx.z;
    const int m0 = blockIdx.y * 32;
    const int n0 = blockIdx.x * 64;
    const int cnt = counts[e];
    if (m0 >= cnt) return;

    __shared__ unsigned short sA[2][32][72];
    __shared__ unsigned short sB[2][64][72];
    __shared__ int   sSlot[32];
    __shared__ float sW[32];

    const int tid  = threadIdx.x;
    const int lane = tid & 31;
    const int wave = tid >> 5;
    const int wm   = wave >> 2;
    const int wn   = wave & 3;

    if (tid < 32) {
        int r = m0 + tid;
        int a = (r < cnt) ? slots[e * CAPQ + r] : -1;
        sSlot[tid] = a;
        sW[tid] = (a >= 0) ? topk_w[a] : 0.0f;
    }
    __syncthreads();

    const int rA = tid >> 3;
    const int cA = (tid & 7) * 8;
    const int slotA = sSlot[rA];
    const unsigned short* aSrc =
        hbuf + (size_t)((slotA >= 0) ? slotA : 0) * IDIM + cA;

    const unsigned short* wd = dwnb + (size_t)e * HDIM * IDIM;
    const unsigned short* bSrc[2];
    int bR[2], bC[2];
#pragma unroll
    for (int i = 0; i < 2; ++i) {
        int f  = tid + 256 * i;           // 0..511
        bR[i]  = f >> 3;                  // 0..63
        bC[i]  = (f & 7) * 8;
        bSrc[i] = wd + (size_t)(n0 + bR[i]) * IDIM + bC[i];
    }

    // prologue: stage chunk 0
    {
        *(uint4*)&sA[0][rA][cA] = *(const uint4*)aSrc;
#pragma unroll
        for (int i = 0; i < 2; ++i)
            *(uint4*)&sB[0][bR[i]][bC[i]] = *(const uint4*)bSrc[i];
    }
    __syncthreads();

    v8f acc = {};
    const int kb   = (lane >> 4) * 8;
    const int mrow = wm * 16 + (lane & 15);
    const int nrow = wn * 16 + (lane & 15);

    int cur = 0;
    for (int k0 = 0; k0 < IDIM; k0 += 64) {
        const int kn = k0 + 64;
        const bool more = kn < IDIM;
        uint4 ra, rb0, rb1;
        if (more) {
            ra  = *(const uint4*)(aSrc + kn);
            rb0 = *(const uint4*)(bSrc[0] + kn);
            rb1 = *(const uint4*)(bSrc[1] + kn);
        }
#pragma unroll
        for (int kk = 0; kk < 64; kk += 32) {
            Frag fa, fb;
            fa.q[0] = *(const uint4*)&sA[cur][mrow][kk + kb];
            fa.q[1] = *(const uint4*)&sA[cur][mrow][kk + kb + 16];
            fb.q[0] = *(const uint4*)&sB[cur][nrow][kk + kb];
            fb.q[1] = *(const uint4*)&sB[cur][nrow][kk + kb + 16];
            acc = __builtin_amdgcn_wmma_f32_16x16x32_bf16(
                false, fa.v, false, fb.v, (short)0, acc, false, false);
        }
        if (more) {
            int nxt = cur ^ 1;
            *(uint4*)&sA[nxt][rA][cA] = ra;
            *(uint4*)&sB[nxt][bR[0]][bC[0]] = rb0;
            *(uint4*)&sB[nxt][bR[1]][bC[1]] = rb1;
        }
        __syncthreads();
        cur ^= 1;
    }

    // epilogue: scale by routing weight and scatter-add into y
    const int ncol = n0 + wn * 16 + (lane & 15);
    const int mhi  = (lane >> 4) * 8;
#pragma unroll
    for (int r = 0; r < 8; ++r) {
        int mi = wm * 16 + mhi + r;
        if (m0 + mi < cnt) {
            int a = sSlot[mi];
            int t = a >> 2;               // TOPK == 4
            atomicAdd(&y[(size_t)t * HDIM + ncol], acc[r] * sW[mi]);
        }
    }
}

// --------------------------------------------------------------------- launch
extern "C" void kernel_launch(void* const* d_in, const int* in_sizes, int n_in,
                              void* d_out, int out_size, void* d_ws, size_t ws_size,
                              hipStream_t stream) {
    const float* hs   = (const float*)d_in[0];   // hidden_states [T,H]
    const int*   tki  = (const int*)  d_in[1];   // top_k_index   [T,K]
    const float* tkw  = (const float*)d_in[2];   // top_k_weights [T,K]
    const float* gup  = (const float*)d_in[3];   // gate_up_proj  [E,2I,H]
    const float* dwn  = (const float*)d_in[4];   // down_proj     [E,H,I]
    float*       y    = (float*)d_out;           // [T,H]

    // workspace layout (bytes)
    char* ws = (char*)d_ws;
    size_t off = 0;
    int* counts = (int*)(ws + off);                 off += 512;
    int* slots  = (int*)(ws + off);                 off += (size_t)E_EXPERTS * CAPQ * 4;
    unsigned short* hbuf = (unsigned short*)(ws + off);
    off += (size_t)NASSIGN * IDIM * 2;
    unsigned short* hsb  = (unsigned short*)(ws + off);
    off += (size_t)TTOK * HDIM * 2;
    unsigned short* gupb = (unsigned short*)(ws + off);
    off += (size_t)E_EXPERTS * 2 * IDIM * HDIM * 2;
    unsigned short* dwnb = (unsigned short*)(ws + off);
    off += (size_t)E_EXPERTS * HDIM * IDIM * 2;   // total ~184 MB

    zero_i32_kernel<<<1, 32, 0, stream>>>(counts, E_EXPERTS);
    zero_f32_kernel<<<1024, 256, 0, stream>>>(y, TTOK * HDIM);
    route_scatter_kernel<<<NASSIGN / 256, 256, 0, stream>>>(tki, counts, slots);

    // one-shot fp32 -> bf16 preconversion of activations + weights
    cvt_f32_bf16_kernel<<<2048, 256, 0, stream>>>(
        hs, (unsigned int*)hsb, TTOK * HDIM / 4);
    cvt_f32_bf16_kernel<<<8192, 256, 0, stream>>>(
        gup, (unsigned int*)gupb, E_EXPERTS * 2 * IDIM * HDIM / 4);
    cvt_f32_bf16_kernel<<<8192, 256, 0, stream>>>(
        dwn, (unsigned int*)dwnb, E_EXPERTS * HDIM * IDIM / 4);

    dim3 gA(IDIM / 64, CAPQ / 32, E_EXPERTS);    // (12, 32, 32)
    moe_gateup_kernel<<<gA, 256, 0, stream>>>(hsb, gupb, counts, slots, hbuf);

    dim3 gB(HDIM / 64, CAPQ / 32, E_EXPERTS);    // (16, 32, 32)
    moe_down_kernel<<<gB, 256, 0, stream>>>(hbuf, dwnb, tkw, counts, slots, y);
}
